// Reltok_88476326297842
// MI455X (gfx1250) — matrix operand order repaired and checked by reading, not compiled, not run on hardware
//
#include <hip/hip_runtime.h>

typedef _Float16 f16_t;
typedef __attribute__((ext_vector_type(16))) _Float16 v16h;
typedef __attribute__((ext_vector_type(8)))  _Float16 v8h;
typedef __attribute__((ext_vector_type(8)))  float    v8f;

#define LN_EPS 1e-5f

// ---------------------------------------------------------------------------
// Weight prep: BN folding, Wtop+Wbot fold, fp32 (K,N) -> fp16 (N,K) transpose
// ---------------------------------------------------------------------------
__global__ void k_bnprep(const float* __restrict__ g, const float* __restrict__ b,
                         const float* __restrict__ m, const float* __restrict__ v,
                         float* __restrict__ scale, float* __restrict__ shift) {
  int d = threadIdx.x;
  float s = g[d] * rsqrtf(v[d] + LN_EPS);
  scale[d] = s;
  shift[d] = b[d] - m[d] * s;
}

__global__ void k_wsumT(const float* __restrict__ proj_w, f16_t* __restrict__ out) {
  int idx = blockIdx.x * 256 + threadIdx.x;     // 128*128
  if (idx >= 128 * 128) return;
  int k = idx >> 7, n = idx & 127;
  out[n * 128 + k] = (f16_t)(proj_w[k * 128 + n] + proj_w[(128 + k) * 128 + n]);
}

__global__ void k_convT(const float* __restrict__ in, f16_t* __restrict__ out, int K, int N) {
  int idx = blockIdx.x * 256 + threadIdx.x;
  if (idx >= K * N) return;
  int k = idx / N, n = idx % N;
  out[(size_t)n * K + k] = (f16_t)in[idx];
}

// ---------------------------------------------------------------------------
// Masked row-reduction over rel_matrix: s2[b,i,0:2] = sum_j mask_j * rel[b,i,j,:]
// and per-batch valid count (written once by the i==0 block).
// ---------------------------------------------------------------------------
__global__ __launch_bounds__(256)
void k_relreduce(const float* __restrict__ rel, const unsigned char* __restrict__ mask,
                 float* __restrict__ s2, float* __restrict__ cntf, int N) {
  int bi = blockIdx.x;               // b*N + i
  int b  = bi / N;
  int i  = bi % N;
  const float* row = rel + (size_t)bi * N * 2;
  const unsigned char* mrow = mask + (size_t)b * N;
  float a0 = 0.f, a1 = 0.f, c = 0.f;
  for (int j = threadIdx.x; j < N; j += 256) {
    float m = mrow[j] ? 1.f : 0.f;
    a0 += m * row[2 * j];
    a1 += m * row[2 * j + 1];
    c  += m;
  }
  __shared__ float r0[256], r1[256], r2[256];
  int t = threadIdx.x;
  r0[t] = a0; r1[t] = a1; r2[t] = c;
  __syncthreads();
  for (int s = 128; s > 0; s >>= 1) {
    if (t < s) { r0[t] += r0[t + s]; r1[t] += r1[t + s]; r2[t] += r2[t + s]; }
    __syncthreads();
  }
  if (t == 0) {
    s2[(size_t)bi * 2]     = r0[0];
    s2[(size_t)bi * 2 + 1] = r1[0];
    if (i == 0) cntf[b] = r2[0];
  }
}

// pooled[b,i,d] = mask_i ? (s0*pre_w[0,d] + s1*pre_w[1,d] + M*pre_b[d]) / max(M,1) : 0
__global__ __launch_bounds__(128)
void k_pooled(const float* __restrict__ s2, const float* __restrict__ cntf,
              const unsigned char* __restrict__ mask,
              const float* __restrict__ pre_w, const float* __restrict__ pre_b,
              f16_t* __restrict__ pooled, int N, int D) {
  int tok = blockIdx.x;              // b*N + i
  int b = tok / N;
  int d = threadIdx.x;
  float out = 0.f;
  if (mask[tok]) {
    float M  = cntf[b];
    float s0 = s2[(size_t)tok * 2];
    float s1 = s2[(size_t)tok * 2 + 1];
    float denom = (M > 1.f) ? M : 1.f;
    out = (s0 * pre_w[d] + s1 * pre_w[D + d] + M * pre_b[d]) / denom;
  }
  pooled[(size_t)tok * D + d] = (f16_t)out;
}

// ---------------------------------------------------------------------------
// Generic batched WMMA GEMM:  C[m,n] = epilogue( alpha * sum_k A[m,k]*Bt[n,k] )
// block = 256 threads = 8 waves; each wave computes a 16x64 tile.
// K-loop is software-pipelined with register ping-pong (K % 64 == 0) so the
// global_load_b128 fragment loads for the next chunk overlap the 4-WMMA burst
// of the current chunk instead of stalling on s_wait_loadcnt 0.
// Fragment layouts (A and B share a contiguous k order, not the ISA table order; C as in the ISA):
//  A (16x32 f16): lane L -> row (L&15); K = k0 + (L&16?16:0) + 0..15 contiguous.
//  B (32x16 f16) from (N,K) row-major Bt: lane L -> col (L&15);
//     K = k0 + (L&16?16:0) + 0..15 contiguous (one 32B load).
//  C f32: VGPR r -> row mT*16 + r + (L&16?8:0), col n + (L&15)
// ---------------------------------------------------------------------------
__global__ __launch_bounds__(256)
void k_gemm(const f16_t* __restrict__ A, int lda, long long sA,
            const f16_t* __restrict__ Bt, int ldb, long long sB,
            int M, int N, int K, float alpha,
            const float* __restrict__ bias,
            const unsigned char* __restrict__ rowmask, long long sRM,
            const float* __restrict__ scale, const float* __restrict__ shift,
            int do_relu,
            const float* __restrict__ residual, long long sRes,
            const float* __restrict__ addrows, int addld,
            const int* __restrict__ rowidx, long long sIdx,
            float* __restrict__ C32, long long sC32,
            f16_t* __restrict__ C16, long long sC16,
            f16_t* __restrict__ Ct16, int ldct, long long sCt) {
  int b = blockIdx.y;
  A  += (long long)b * sA;
  Bt += (long long)b * sB;
  if (C32)      C32      += (long long)b * sC32;
  if (C16)      C16      += (long long)b * sC16;
  if (Ct16)     Ct16     += (long long)b * sCt;
  if (residual) residual += (long long)b * sRes;
  if (rowmask)  rowmask  += (long long)b * sRM;
  if (rowidx)   rowidx   += (long long)b * sIdx;

  int wid  = threadIdx.x >> 5;
  int lane = threadIdx.x & 31;
  int nT   = N >> 6;                          // 64-wide n tiles
  int wt   = blockIdx.x * 8 + wid;
  if (wt >= (M >> 4) * nT) return;            // uniform per wave (EXEC stays all-1s)
  int mT  = wt / nT;
  int n64 = wt % nT;

  int hi   = (lane >> 4) & 1;
  int rowA = mT * 16 + (lane & 15);
  const f16_t* Ap  = A + (size_t)rowA * lda + hi * 16;
  int nb0 = n64 * 64 + (lane & 15);
  const f16_t* Bp0 = Bt + (size_t)(nb0 +  0) * ldb + hi * 16;
  const f16_t* Bp1 = Bt + (size_t)(nb0 + 16) * ldb + hi * 16;
  const f16_t* Bp2 = Bt + (size_t)(nb0 + 32) * ldb + hi * 16;
  const f16_t* Bp3 = Bt + (size_t)(nb0 + 48) * ldb + hi * 16;

  auto loadA = [&](int off) -> v16h {
    v8h lo  = *(const v8h*)(Ap + off);
    v8h hi8 = *(const v8h*)(Ap + off + 8);
    v16h r;
#pragma unroll
    for (int t = 0; t < 8; ++t) { r[t] = lo[t]; r[8 + t] = hi8[t]; }
    return r;
  };

  v8f zero = {};
  v8f acc[4];
#pragma unroll
  for (int s = 0; s < 4; ++s) acc[s] = zero;

  // prologue: ping buffers for k = 0
  v16h aP  = loadA(0);
  v16h bP0 = *(const v16h*)(Bp0);
  v16h bP1 = *(const v16h*)(Bp1);
  v16h bP2 = *(const v16h*)(Bp2);
  v16h bP3 = *(const v16h*)(Bp3);

  for (int k0 = 0; k0 < K; k0 += 64) {
    int k1 = k0 + 32;
    // pong loads (k1) issued before consuming ping (k0)
    v16h aQ  = loadA(k1);
    v16h bQ0 = *(const v16h*)(Bp0 + k1);
    v16h bQ1 = *(const v16h*)(Bp1 + k1);
    v16h bQ2 = *(const v16h*)(Bp2 + k1);
    v16h bQ3 = *(const v16h*)(Bp3 + k1);

    acc[0] = __builtin_amdgcn_wmma_f32_16x16x32_f16(false, aP, false, bP0, (short)0, acc[0], false, false);
    acc[1] = __builtin_amdgcn_wmma_f32_16x16x32_f16(false, aP, false, bP1, (short)0, acc[1], false, false);
    acc[2] = __builtin_amdgcn_wmma_f32_16x16x32_f16(false, aP, false, bP2, (short)0, acc[2], false, false);
    acc[3] = __builtin_amdgcn_wmma_f32_16x16x32_f16(false, aP, false, bP3, (short)0, acc[3], false, false);

    int k2 = k0 + 64;
    if (k2 < K) {
      // refill ping (k2) while pong WMMAs execute
      __builtin_prefetch((const void*)(Ap + k2 + 64), 0, 1);
      aP  = loadA(k2);
      bP0 = *(const v16h*)(Bp0 + k2);
      bP1 = *(const v16h*)(Bp1 + k2);
      bP2 = *(const v16h*)(Bp2 + k2);
      bP3 = *(const v16h*)(Bp3 + k2);
    }

    acc[0] = __builtin_amdgcn_wmma_f32_16x16x32_f16(false, aQ, false, bQ0, (short)0, acc[0], false, false);
    acc[1] = __builtin_amdgcn_wmma_f32_16x16x32_f16(false, aQ, false, bQ1, (short)0, acc[1], false, false);
    acc[2] = __builtin_amdgcn_wmma_f32_16x16x32_f16(false, aQ, false, bQ2, (short)0, acc[2], false, false);
    acc[3] = __builtin_amdgcn_wmma_f32_16x16x32_f16(false, aQ, false, bQ3, (short)0, acc[3], false, false);
  }

  int mbase = mT * 16 + hi * 8;
  int ncol  = n64 * 64 + (lane & 15);
#pragma unroll
  for (int r = 0; r < 8; ++r) {
    int m = mbase + r;
    float rm = 1.f;
    if (rowmask) rm = rowmask[m] ? 1.f : 0.f;
    int gidx = rowidx ? rowidx[m] : 0;
#pragma unroll
    for (int s = 0; s < 4; ++s) {
      int n = ncol + s * 16;
      float v = acc[s][r] * alpha;
      if (bias)     v += bias[n] * rm;
      if (scale)    v  = v * scale[n] + shift[n];
      if (do_relu)  v  = v > 0.f ? v : 0.f;
      if (residual) v += residual[(size_t)m * N + n];
      if (addrows)  v += addrows[(size_t)gidx * addld + n];
      if (C32)  C32[(size_t)m * N + n] = v;
      if (C16)  C16[(size_t)m * N + n] = (f16_t)v;
      if (Ct16) Ct16[(size_t)n * ldct + m] = (f16_t)v;
    }
  }
}

// ---------------------------------------------------------------------------
// Row softmax over 512 columns with key-padding bias; fp32 in, fp16 out
// ---------------------------------------------------------------------------
__global__ __launch_bounds__(256)
void k_softmax(const float* __restrict__ S, const unsigned char* __restrict__ mask,
               f16_t* __restrict__ P, int N) {
  int row = blockIdx.x;              // b*N + i
  int b   = row / N;
  const float* Sr = S + (size_t)row * N;
  const unsigned char* mr = mask + (size_t)b * N;
  f16_t* Pr = P + (size_t)row * N;
  int t = threadIdx.x;
  float v0 = Sr[t]       + (mr[t]       ? 0.f : -1e9f);
  float v1 = Sr[t + 256] + (mr[t + 256] ? 0.f : -1e9f);
  __shared__ float red[256];
  red[t] = fmaxf(v0, v1);
  __syncthreads();
  for (int s = 128; s > 0; s >>= 1) { if (t < s) red[t] = fmaxf(red[t], red[t + s]); __syncthreads(); }
  float gmax = red[0];
  __syncthreads();
  float e0 = __expf(v0 - gmax), e1 = __expf(v1 - gmax);
  red[t] = e0 + e1;
  __syncthreads();
  for (int s = 128; s > 0; s >>= 1) { if (t < s) red[t] += red[t + s]; __syncthreads(); }
  float inv = 1.f / red[0];
  Pr[t]       = (f16_t)(e0 * inv);
  Pr[t + 256] = (f16_t)(e1 * inv);
}

// ---------------------------------------------------------------------------
// LayerNorm over D=128 per token; writes fp32 (residual stream) + fp16 mirror
// ---------------------------------------------------------------------------
__global__ __launch_bounds__(128)
void k_layernorm(const float* __restrict__ Y, const float* __restrict__ g,
                 const float* __restrict__ be, float* __restrict__ O32,
                 f16_t* __restrict__ O16) {
  int tok = blockIdx.x;
  int d = threadIdx.x;
  float v = Y[(size_t)tok * 128 + d];
  __shared__ float red[128];
  red[d] = v;
  __syncthreads();
  for (int s = 64; s > 0; s >>= 1) { if (d < s) red[d] += red[d + s]; __syncthreads(); }
  float mean = red[0] * (1.f / 128.f);
  __syncthreads();
  float c = v - mean;
  red[d] = c * c;
  __syncthreads();
  for (int s = 64; s > 0; s >>= 1) { if (d < s) red[d] += red[d + s]; __syncthreads(); }
  float var = red[0] * (1.f / 128.f);
  float o = c * rsqrtf(var + LN_EPS) * g[d] + be[d];
  O32[(size_t)tok * 128 + d] = o;
  O16[(size_t)tok * 128 + d] = (f16_t)o;
}

// ---------------------------------------------------------------------------
extern "C" void kernel_launch(void* const* d_in, const int* in_sizes, int n_in,
                              void* d_out, int out_size, void* d_ws, size_t ws_size,
                              hipStream_t stream) {
  (void)in_sizes; (void)n_in; (void)out_size; (void)ws_size;
  const float* rel    = (const float*)d_in[0];
  const float* pre_w  = (const float*)d_in[1];
  const float* pre_b  = (const float*)d_in[2];
  const float* proj_w = (const float*)d_in[3];
  const float* proj_b = (const float*)d_in[4];
  const float* out_w1 = (const float*)d_in[5];
  const float* bn_g   = (const float*)d_in[6];
  const float* bn_b   = (const float*)d_in[7];
  const float* bn_m   = (const float*)d_in[8];
  const float* bn_v   = (const float*)d_in[9];
  const float* out_w2 = (const float*)d_in[10];
  const float* out_b2 = (const float*)d_in[11];
  const float* pe     = (const float*)d_in[12];
  const float* wq     = (const float*)d_in[13];
  const float* bq     = (const float*)d_in[14];
  const float* wk     = (const float*)d_in[15];
  const float* bk     = (const float*)d_in[16];
  const float* wv     = (const float*)d_in[17];
  const float* bv     = (const float*)d_in[18];
  const float* wo     = (const float*)d_in[19];
  const float* bo     = (const float*)d_in[20];
  const float* ln1g   = (const float*)d_in[21];
  const float* ln1b   = (const float*)d_in[22];
  const float* ln2g   = (const float*)d_in[23];
  const float* ln2b   = (const float*)d_in[24];
  const float* fw1    = (const float*)d_in[25];
  const float* fb1    = (const float*)d_in[26];
  const float* fw2    = (const float*)d_in[27];
  const float* fb2    = (const float*)d_in[28];
  const unsigned char* mask = (const unsigned char*)d_in[29];
  const int* agent_id       = (const int*)d_in[30];

  const int N = 512, D = 128;
  const long long TD = 512LL * 128;   // per-batch token-major stride
  const long long SS = 512LL * 512;   // per-batch attention-matrix stride

  char* wp = (char*)d_ws;
  auto alloc = [&](size_t bytes) -> void* {
    void* r = (void*)wp;
    wp += (bytes + 255) & ~(size_t)255;
    return r;
  };
  f16_t* wsumT = (f16_t*)alloc(128 * 128 * 2);
  f16_t* w1T   = (f16_t*)alloc(128 * 128 * 2);
  f16_t* w2T   = (f16_t*)alloc(128 * 128 * 2);
  f16_t* wqT   = (f16_t*)alloc(3 * 128 * 128 * 2);
  f16_t* wkT   = (f16_t*)alloc(3 * 128 * 128 * 2);
  f16_t* wvT   = (f16_t*)alloc(3 * 128 * 128 * 2);
  f16_t* woT   = (f16_t*)alloc(3 * 128 * 128 * 2);
  f16_t* fw1T  = (f16_t*)alloc(3 * 128 * 512 * 2);
  f16_t* fw2T  = (f16_t*)alloc(3 * 512 * 128 * 2);
  float* bnscale = (float*)alloc(128 * 4);
  float* bnshift = (float*)alloc(128 * 4);
  float* s2      = (float*)alloc(1024 * 2 * 4);
  float* cntf    = (float*)alloc(2 * 4);
  f16_t* pooledb = (f16_t*)alloc(1024 * 128 * 2);
  f16_t* X1b     = (f16_t*)alloc(1024 * 128 * 2);
  f16_t* Hb      = (f16_t*)alloc(1024 * 128 * 2);
  float* xf32    = (float*)alloc(1024 * 128 * 4);
  f16_t* xf16    = (f16_t*)alloc(1024 * 128 * 2);
  f16_t* Qb      = (f16_t*)alloc(1024 * 128 * 2);
  f16_t* Kb      = (f16_t*)alloc(1024 * 128 * 2);
  f16_t* Vtb     = (f16_t*)alloc(1024 * 128 * 2);   // (B,128,512) transposed V
  float* Sb      = (float*)alloc(2 * 512 * 512 * 4);
  f16_t* Pb      = (f16_t*)alloc(2 * 512 * 512 * 2);
  f16_t* attb    = (f16_t*)alloc(1024 * 128 * 2);
  float* yb      = (float*)alloc(1024 * 128 * 4);
  f16_t* h1b     = (f16_t*)alloc(2 * 512 * 512 * 2);

  // ---- weight prep (idempotent each launch) ----
  k_bnprep<<<1, 128, 0, stream>>>(bn_g, bn_b, bn_m, bn_v, bnscale, bnshift);
  k_wsumT<<<64, 256, 0, stream>>>(proj_w, wsumT);
  k_convT<<<64, 256, 0, stream>>>(out_w1, w1T, 128, 128);
  k_convT<<<64, 256, 0, stream>>>(out_w2, w2T, 128, 128);
  for (int l = 0; l < 3; ++l) {
    k_convT<<<64, 256, 0, stream>>>(wq + l * 16384, wqT + l * 16384, 128, 128);
    k_convT<<<64, 256, 0, stream>>>(wk + l * 16384, wkT + l * 16384, 128, 128);
    k_convT<<<64, 256, 0, stream>>>(wv + l * 16384, wvT + l * 16384, 128, 128);
    k_convT<<<64, 256, 0, stream>>>(wo + l * 16384, woT + l * 16384, 128, 128);
    k_convT<<<256, 256, 0, stream>>>(fw1 + l * 65536, fw1T + l * 65536, 128, 512);
    k_convT<<<256, 256, 0, stream>>>(fw2 + l * 65536, fw2T + l * 65536, 512, 128);
  }

  // ---- collapsed pairwise front-end ----
  k_relreduce<<<1024, 256, 0, stream>>>(rel, mask, s2, cntf, N);
  k_pooled<<<1024, 128, 0, stream>>>(s2, cntf, mask, pre_w, pre_b, pooledb, N, D);

  auto gemm = [&](const f16_t* A, int lda, long long sA,
                  const f16_t* Bt, int ldb, long long sB,
                  int M, int Nn, int Kk, float alpha,
                  const float* bias, const unsigned char* rmask, long long sRM,
                  const float* sc, const float* sh, int relu,
                  const float* res, long long sRes,
                  const float* addr, int addld, const int* ridx, long long sIdx,
                  float* C32, long long sC32, f16_t* C16, long long sC16,
                  f16_t* Ct, int ldct, long long sCt) {
    int wtiles = (M / 16) * (Nn / 64);
    dim3 grid((wtiles + 7) / 8, 2);
    k_gemm<<<grid, 256, 0, stream>>>(A, lda, sA, Bt, ldb, sB, M, Nn, Kk, alpha,
                                     bias, rmask, sRM, sc, sh, relu, res, sRes,
                                     addr, addld, ridx, sIdx, C32, sC32, C16, sC16,
                                     Ct, ldct, sCt);
  };

  // X1 = pooled @ (Wtop+Wbot) + proj_b  (bias gated by token validity)
  gemm(pooledb, 128, TD, wsumT, 128, 0, 512, 128, 128, 1.f,
       proj_b, mask, 512, nullptr, nullptr, 0,
       nullptr, 0, nullptr, 0, nullptr, 0,
       nullptr, 0, X1b, TD, nullptr, 0, 0);
  // H = relu(bn(X1 @ out_w1))
  gemm(X1b, 128, TD, w1T, 128, 0, 512, 128, 128, 1.f,
       nullptr, nullptr, 0, bnscale, bnshift, 1,
       nullptr, 0, nullptr, 0, nullptr, 0,
       nullptr, 0, Hb, TD, nullptr, 0, 0);
  // x = H @ out_w2 + out_b2 + pe[agent_id]
  gemm(Hb, 128, TD, w2T, 128, 0, 512, 128, 128, 1.f,
       out_b2, nullptr, 0, nullptr, nullptr, 0,
       nullptr, 0, pe, 128, agent_id, 512,
       xf32, TD, xf16, TD, nullptr, 0, 0);

  // ---- 3 post-norm transformer layers ----
  const float attn_scale = 0.08838834764831845f;   // 1/sqrt(128)
  for (int l = 0; l < 3; ++l) {
    gemm(xf16, 128, TD, wqT + l * 16384, 128, 0, 512, 128, 128, 1.f,
         bq + l * 128, nullptr, 0, nullptr, nullptr, 0,
         nullptr, 0, nullptr, 0, nullptr, 0,
         nullptr, 0, Qb, TD, nullptr, 0, 0);
    gemm(xf16, 128, TD, wkT + l * 16384, 128, 0, 512, 128, 128, 1.f,
         bk + l * 128, nullptr, 0, nullptr, nullptr, 0,
         nullptr, 0, nullptr, 0, nullptr, 0,
         nullptr, 0, Kb, TD, nullptr, 0, 0);
    gemm(xf16, 128, TD, wvT + l * 16384, 128, 0, 512, 128, 128, 1.f,
         bv + l * 128, nullptr, 0, nullptr, nullptr, 0,
         nullptr, 0, nullptr, 0, nullptr, 0,
         nullptr, 0, nullptr, 0, Vtb, 512, TD);
    // S = (Q @ K^T) * scale   (Bt == K token-major directly)
    gemm(Qb, 128, TD, Kb, 128, TD, 512, 512, 128, attn_scale,
         nullptr, nullptr, 0, nullptr, nullptr, 0,
         nullptr, 0, nullptr, 0, nullptr, 0,
         Sb, SS, nullptr, 0, nullptr, 0, 0);
    k_softmax<<<1024, 256, 0, stream>>>(Sb, mask, Pb, N);
    // att = P @ V   (Bt == V^T)
    gemm(Pb, 512, SS, Vtb, 512, TD, 512, 128, 512, 1.f,
         nullptr, nullptr, 0, nullptr, nullptr, 0,
         nullptr, 0, nullptr, 0, nullptr, 0,
         nullptr, 0, attb, TD, nullptr, 0, 0);
    // y = att @ wo + bo + x
    gemm(attb, 128, TD, woT + l * 16384, 128, 0, 512, 128, 128, 1.f,
         bo + l * 128, nullptr, 0, nullptr, nullptr, 0,
         xf32, TD, nullptr, 0, nullptr, 0,
         yb, TD, nullptr, 0, nullptr, 0, 0);
    k_layernorm<<<1024, 128, 0, stream>>>(yb, ln1g + l * 128, ln1b + l * 128, xf32, xf16);
    // h1 = relu(x @ fw1 + fb1)
    gemm(xf16, 128, TD, fw1T + l * 65536, 128, 0, 512, 512, 128, 1.f,
         fb1 + l * 512, nullptr, 0, nullptr, nullptr, 1,
         nullptr, 0, nullptr, 0, nullptr, 0,
         nullptr, 0, h1b, SS, nullptr, 0, 0);
    // y = h1 @ fw2 + fb2 + x
    gemm(h1b, 512, SS, fw2T + l * 65536, 512, 0, 512, 128, 512, 1.f,
         fb2 + l * 128, nullptr, 0, nullptr, nullptr, 0,
         xf32, TD, nullptr, 0, nullptr, 0,
         yb, TD, nullptr, 0, nullptr, 0, 0);
    float* lnout = (l == 2) ? (float*)d_out : xf32;
    k_layernorm<<<1024, 128, 0, stream>>>(yb, ln2g + l * 128, ln2b + l * 128, lnout, xf16);
  }
}
